// CrossDeformableAttentionBlock_82317343195281
// MI455X (gfx1250) — matrix operand order repaired and checked by reading, not compiled, not run on hardware
//
#include <hip/hip_runtime.h>
#include <hip/hip_bf16.h>

#define BB 2
#define CC 96
#define HH 128
#define WW 128
#define PP (HH * WW)          // 16384
#define GG 12
#define NHH 12
#define KK 9
#define CGg 8
#define HDD 8
#define MLPC 192

typedef __attribute__((ext_vector_type(16))) __bf16 v16bf;
typedef __attribute__((ext_vector_type(8)))  float  v8f;
typedef unsigned short ushort_t;

union BF16x16 {
    v16bf v;
    ushort_t u[16];
    uint4 q[2];
};

__device__ __forceinline__ ushort_t f32_to_bf16(float f) {
    union { float f; unsigned u; } a; a.f = f;
    unsigned r = a.u + 0x7FFFu + ((a.u >> 16) & 1u);
    return (ushort_t)(r >> 16);
}
__device__ __forceinline__ float bf16_to_f32(ushort_t h) {
    union { unsigned u; float f; } a; a.u = ((unsigned)h) << 16;
    return a.f;
}
__device__ __forceinline__ float gelu_tanh(float x) {
    float x3 = x * x * x;
    return 0.5f * x * (1.0f + tanhf(0.7978845608f * (x + 0.044715f * x3)));
}

// ---------------------------------------------------------------------------
// Weight packing: f32 [OC][IC](x NTAP taps) -> bf16 packed A-fragments
// packed[(((tap*MT+mt)*NC+ch)*32+lane)*16+j]
//   = W[mt*16+(lane&15)][ch*32 + ((lane>>4)*16) + j]
// ---------------------------------------------------------------------------
__global__ void pack_w_kernel(const float* __restrict__ src, ushort_t* __restrict__ dst,
                              int OC, int IC, int ICpad, int MT, int NTAP) {
    int NC = ICpad / 32;
    long total = (long)NTAP * MT * NC * 32 * 16;
    for (long i = (long)blockIdx.x * blockDim.x + threadIdx.x; i < total;
         i += (long)gridDim.x * blockDim.x) {
        int j = (int)(i & 15); long t = i >> 4;
        int lane = (int)(t & 31); t >>= 5;
        int ch = (int)(t % NC); t /= NC;
        int mt = (int)(t % MT); int tap = (int)(t / MT);
        int m  = mt * 16 + (lane & 15);
        int kk = ((lane >> 4) * 16) + j;
        int ic = ch * 32 + kk;
        float v = 0.f;
        if (m < OC && ic < IC) v = src[((long)m * IC + ic) * NTAP + tap];
        dst[i] = f32_to_bf16(v);
    }
}

// ---------------------------------------------------------------------------
// Bilinear helpers (zero padding outside, as in reference)
// ---------------------------------------------------------------------------
struct Corner { int idx; float w; };

__device__ __forceinline__ void bilin_setup(float xs, float ys, Corner cr[4]) {
    float x0f = floorf(xs), y0f = floorf(ys);
    int x0 = (int)x0f, y0 = (int)y0f;
    float wx = xs - x0f, wy = ys - y0f;
    float ww[4] = { (1.f - wx) * (1.f - wy), wx * (1.f - wy),
                    (1.f - wx) * wy,         wx * wy };
    const int dxs[4] = {0, 1, 0, 1};
    const int dys[4] = {0, 0, 1, 1};
#pragma unroll
    for (int i = 0; i < 4; ++i) {
        int xi = x0 + dxs[i], yi = y0 + dys[i];
        bool valid = (xi >= 0) && (xi < WW) && (yi >= 0) && (yi < HH);
        int xc = min(max(xi, 0), WW - 1);
        int yc = min(max(yi, 0), HH - 1);
        cr[i].idx = yc * WW + xc;
        cr[i].w = valid ? ww[i] : 0.f;
    }
}
__device__ __forceinline__ float bilin_apply(const float* __restrict__ img, const Corner cr[4]) {
    return img[cr[0].idx] * cr[0].w + img[cr[1].idx] * cr[1].w +
           img[cr[2].idx] * cr[2].w + img[cr[3].idx] * cr[3].w;
}

// ---------------------------------------------------------------------------
// fill_x1: conv1 input x1 bf16 [B][224][P]: ch 0..95 = feat_t, 192/193 = flow,
// 194..223 = 0.  Also writes xt_bf = bf16 copy of feat_t [B][96][P].
// ---------------------------------------------------------------------------
__global__ void fill_x1_kernel(const float* __restrict__ feat_t, const float* __restrict__ flow,
                               ushort_t* __restrict__ x1, ushort_t* __restrict__ xt_bf) {
    long i = (long)blockIdx.x * blockDim.x + threadIdx.x;
    if (i >= (long)BB * PP) return;
    int b = (int)(i / PP), p = (int)(i % PP);
    for (int c = 0; c < CC; ++c) {
        ushort_t u = f32_to_bf16(feat_t[((long)b * CC + c) * PP + p]);
        x1[((long)b * 224 + c) * PP + p] = u;
        xt_bf[((long)b * CC + c) * PP + p] = u;
    }
    x1[((long)b * 224 + 192) * PP + p] = f32_to_bf16(flow[((long)b * 2 + 0) * PP + p]);
    x1[((long)b * 224 + 193) * PP + p] = f32_to_bf16(flow[((long)b * 2 + 1) * PP + p]);
    for (int c = 194; c < 224; ++c) x1[((long)b * 224 + c) * PP + p] = 0;
}

// ---------------------------------------------------------------------------
// Flow warp: kv_to_q written into x1 channels 96..191 (bf16)
// ---------------------------------------------------------------------------
__global__ void warp_bilinear_kernel(const float* __restrict__ feat_kv,
                                     const float* __restrict__ flow,
                                     ushort_t* __restrict__ x1) {
    long i = (long)blockIdx.x * blockDim.x + threadIdx.x;
    if (i >= (long)BB * PP) return;
    int b = (int)(i / PP), p = (int)(i % PP);
    int y = p / WW, x = p % WW;
    float xs = (float)x + flow[((long)b * 2 + 0) * PP + p];
    float ys = (float)y + flow[((long)b * 2 + 1) * PP + p];
    Corner cr[4];
    bilin_setup(xs, ys, cr);
    for (int c = 0; c < CC; ++c) {
        const float* img = feat_kv + ((long)b * CC + c) * PP;
        x1[((long)b * 224 + 96 + c) * PP + p] = f32_to_bf16(bilin_apply(img, cr));
    }
}

// ---------------------------------------------------------------------------
// Straight-line compute cores (templated on #M-tiles per wave).
// All four B fragments are preloaded before the WMMA burst so the scheduler
// can clause the loads, take one wait, then run 4/8 back-to-back WMMAs.
// ---------------------------------------------------------------------------
__device__ __forceinline__ v8f wmma_bf16(const BF16x16& a, const BF16x16& b, v8f c) {
    return __builtin_amdgcn_wmma_f32_16x16x32_bf16(
        false, a.v, false, b.v, (short)0, c, false, false);
}

template<bool TWO>
__device__ __forceinline__ void conv_core(const ushort_t* __restrict__ lds,
                                          const ushort_t* __restrict__ wpk,
                                          int NC, int LROW, int MT,
                                          int mt0, int mt1, int lane, int nb, int kb2,
                                          v8f acc0[4], v8f acc1[4]) {
    const int XW = 66;
    for (int tap = 0; tap < 9; ++tap) {
        int dy = tap / 3, dx = tap % 3;
        const ushort_t* wt = wpk + (long)tap * MT * NC * 512;
        for (int ch = 0; ch < NC; ++ch) {
            BF16x16 a0, a1;
            {
                const ushort_t* s = wt + (((long)mt0 * NC + ch) * 32 + lane) * 16;
                a0.q[0] = *(const uint4*)s; a0.q[1] = *(const uint4*)(s + 8);
            }
            if (TWO) {
                const ushort_t* s = wt + (((long)mt1 * NC + ch) * 32 + lane) * 16;
                a1.q[0] = *(const uint4*)s; a1.q[1] = *(const uint4*)(s + 8);
            }
            BF16x16 bf[4];
#pragma unroll
            for (int t = 0; t < 4; ++t) {
                const ushort_t* bsrc =
                    &lds[(dy * XW + nb + 16 * t + dx) * LROW + ch * 32 + kb2];
                bf[t].q[0] = *(const uint4*)(bsrc);
                bf[t].q[1] = *(const uint4*)(bsrc + 8);
            }
#pragma unroll
            for (int t = 0; t < 4; ++t) {
                acc0[t] = wmma_bf16(a0, bf[t], acc0[t]);
                if (TWO) acc1[t] = wmma_bf16(a1, bf[t], acc1[t]);
            }
        }
    }
}

template<bool TWO>
__device__ __forceinline__ void gemm_core(const ushort_t* __restrict__ lds,
                                          const ushort_t* __restrict__ wpk,
                                          int NC, int LROW,
                                          int mt0, int mt1, int lane, int nb, int kb2,
                                          v8f acc0[4], v8f acc1[4]) {
    for (int ch = 0; ch < NC; ++ch) {
        BF16x16 a0, a1;
        {
            const ushort_t* s = wpk + (((long)mt0 * NC + ch) * 32 + lane) * 16;
            a0.q[0] = *(const uint4*)s; a0.q[1] = *(const uint4*)(s + 8);
        }
        if (TWO) {
            const ushort_t* s = wpk + (((long)mt1 * NC + ch) * 32 + lane) * 16;
            a1.q[0] = *(const uint4*)s; a1.q[1] = *(const uint4*)(s + 8);
        }
        BF16x16 bf[4];
#pragma unroll
        for (int t = 0; t < 4; ++t) {
            const ushort_t* bsrc = &lds[(nb + 16 * t) * LROW + ch * 32 + kb2];
            bf[t].q[0] = *(const uint4*)(bsrc);
            bf[t].q[1] = *(const uint4*)(bsrc + 8);
        }
#pragma unroll
        for (int t = 0; t < 4; ++t) {
            acc0[t] = wmma_bf16(a0, bf[t], acc0[t]);
            if (TWO) acc1[t] = wmma_bf16(a1, bf[t], acc1[t]);
        }
    }
}

// ---------------------------------------------------------------------------
// Implicit-GEMM 3x3 conv via WMMA bf16.
// Block = 256 thr (8 waves), one (b, y, 64-px) tile, all MT oc-tiles.
// LDS: patch [3 rows][66 px][ICpad+8 ch] bf16.
// ---------------------------------------------------------------------------
__global__ void __launch_bounds__(256)
conv3x3_wmma_kernel(const ushort_t* __restrict__ in,   // [B][ICpad][P] bf16
                    const ushort_t* __restrict__ wpk,  // packed [9][MT][NC][32][16]
                    const float* __restrict__ bias, const float* __restrict__ slope,
                    ushort_t* __restrict__ out_bf, float* __restrict__ out_f32,
                    int ICpad, int MT, int OC) {
    const int NC = ICpad / 32;
    const int LROW = ICpad + 8;
    int blk = blockIdx.x;
    int xt = blk & 1;                  // W/64 == 2
    int y  = (blk >> 1) & (HH - 1);
    int b  = blk >> 8;
    int x0 = xt * 64;

    extern __shared__ ushort_t lds[];  // [3][66][LROW]
    const int XW = 66;
    int tid = threadIdx.x;
    int tot = 3 * XW * ICpad;
    for (int i = tid; i < tot; i += 256) {
        int xi = i % XW; int t = i / XW;
        int c = t % ICpad; int r = t / ICpad;
        int yy = y + r - 1, xx = x0 + xi - 1;
        ushort_t v = 0;
        if ((unsigned)yy < (unsigned)HH && (unsigned)xx < (unsigned)WW)
            v = in[((long)b * ICpad + c) * PP + yy * WW + xx];
        lds[(r * XW + xi) * LROW + c] = v;
    }
    __syncthreads();

    int wave = __builtin_amdgcn_readfirstlane((int)threadIdx.x >> 5);  // scalar
    int lane = tid & 31;
    int nb = lane & 15, kb2 = (lane >> 4) * 16;
    int mt0 = wave, mt1 = wave + 8;
    v8f acc0[4], acc1[4];
#pragma unroll
    for (int t = 0; t < 4; ++t) {
        acc0[t] = (v8f){0, 0, 0, 0, 0, 0, 0, 0};
        acc1[t] = (v8f){0, 0, 0, 0, 0, 0, 0, 0};
    }

    if (mt1 < MT)
        conv_core<true>(lds, wpk, NC, LROW, MT, mt0, mt1, lane, nb, kb2, acc0, acc1);
    else if (mt0 < MT)
        conv_core<false>(lds, wpk, NC, LROW, MT, mt0, mt1, lane, nb, kb2, acc0, acc1);

    int nn = lane & 15;
    int mrow = (lane >> 4) << 3;
#pragma unroll
    for (int half = 0; half < 2; ++half) {
        int mt = half ? mt1 : mt0;
        if (mt >= MT) continue;
#pragma unroll
        for (int t = 0; t < 4; ++t) {
            v8f acc = half ? acc1[t] : acc0[t];
#pragma unroll
            for (int r = 0; r < 8; ++r) {
                int oc = mt * 16 + r + mrow;
                if (oc >= OC) continue;
                float v = acc[r] + (bias ? bias[oc] : 0.f);
                if (slope) v = (v >= 0.f) ? v : slope[oc] * v;
                long idx = ((long)b * OC + oc) * PP + y * WW + x0 + 16 * t + nn;
                if (out_bf)  out_bf[idx]  = f32_to_bf16(v);
                if (out_f32) out_f32[idx] = v;
            }
        }
    }
}

// ---------------------------------------------------------------------------
// Generic WMMA GEMM: out[b][oc][n] = act(W[oc][ic] * in[b][ic][n] + bias + resid)
// Block = 256 thr, 64 pixels, all MT oc-tiles.  act: 0=none, 1=gelu.
// ---------------------------------------------------------------------------
__global__ void __launch_bounds__(256)
gemm_wmma_kernel(const ushort_t* __restrict__ in,   // [B][ICpad][Ntot] bf16
                 const ushort_t* __restrict__ wpk,  // packed [MT][NC][32][16]
                 const float* __restrict__ bias, const float* __restrict__ resid,
                 ushort_t* __restrict__ out_bf, float* __restrict__ out_f32,
                 int ICpad, int MT, int OC, long Ntot, int act) {
    const int NC = ICpad / 32;
    const int LROW = ICpad + 8;
    long nblocks = Ntot >> 6;
    long blk = blockIdx.x;
    long nt = blk % nblocks;
    int b = (int)(blk / nblocks);
    long n0 = nt << 6;

    extern __shared__ ushort_t lds[];  // [64][LROW]
    int tid = threadIdx.x;
    int tot = 64 * ICpad;
    for (int i = tid; i < tot; i += 256) {
        int px = i & 63; int c = i >> 6;
        lds[px * LROW + c] = in[((long)b * ICpad + c) * Ntot + n0 + px];
    }
    __syncthreads();

    int wave = __builtin_amdgcn_readfirstlane((int)threadIdx.x >> 5);  // scalar
    int lane = tid & 31;
    int nb = lane & 15, kb2 = (lane >> 4) * 16;
    int mt0 = wave, mt1 = wave + 8;
    v8f acc0[4], acc1[4];
#pragma unroll
    for (int t = 0; t < 4; ++t) {
        acc0[t] = (v8f){0, 0, 0, 0, 0, 0, 0, 0};
        acc1[t] = (v8f){0, 0, 0, 0, 0, 0, 0, 0};
    }

    if (mt1 < MT)
        gemm_core<true>(lds, wpk, NC, LROW, mt0, mt1, lane, nb, kb2, acc0, acc1);
    else if (mt0 < MT)
        gemm_core<false>(lds, wpk, NC, LROW, mt0, mt1, lane, nb, kb2, acc0, acc1);

    int nn = lane & 15;
    int mrow = (lane >> 4) << 3;
#pragma unroll
    for (int half = 0; half < 2; ++half) {
        int mt = half ? mt1 : mt0;
        if (mt >= MT) continue;
#pragma unroll
        for (int t = 0; t < 4; ++t) {
            v8f acc = half ? acc1[t] : acc0[t];
#pragma unroll
            for (int r = 0; r < 8; ++r) {
                int oc = mt * 16 + r + mrow;
                if (oc >= OC) continue;
                long idx = ((long)b * OC + oc) * Ntot + n0 + 16 * t + nn;
                float v = acc[r] + (bias ? bias[oc] : 0.f);
                if (resid) v += resid[idx];
                if (act == 1) v = gelu_tanh(v);
                if (out_bf)  out_bf[idx]  = f32_to_bf16(v);
                if (out_f32) out_f32[idx] = v;
            }
        }
    }
}

// ---------------------------------------------------------------------------
// Grouped deformable sampling.  offsets = conv3_out + flipped flow.
// kv bf16 layout: [B][C][k*P + p]  (channel stride 9P -> long-N GEMM for k/v)
// ---------------------------------------------------------------------------
__global__ void deform_sample_kernel(const float* __restrict__ feat_kv,
                                     const float* __restrict__ flow,
                                     const float* __restrict__ off,
                                     ushort_t* __restrict__ kv) {
    long i = (long)blockIdx.x * blockDim.x + threadIdx.x;
    if (i >= (long)BB * GG * PP) return;
    int p = (int)(i % PP); long t = i / PP;
    int g = (int)(t % GG); int b = (int)(t / GG);
    int y = p / WW, x = p % WW;
    float fx = flow[((long)b * 2 + 0) * PP + p];
    float fy = flow[((long)b * 2 + 1) * PP + p];
    for (int k = 0; k < KK; ++k) {
        int kh = k / 3, kw = k % 3;
        float dy = off[((long)b * 216 + ((g * KK + k) * 2 + 0)) * PP + p] + fy;
        float dx = off[((long)b * 216 + ((g * KK + k) * 2 + 1)) * PP + p] + fx;
        float sy = (float)y + (float)(kh - 1) + dy;
        float sx = (float)x + (float)(kw - 1) + dx;
        Corner cr[4];
        bilin_setup(sx, sy, cr);
        for (int cg = 0; cg < CGg; ++cg) {
            int c = g * CGg + cg;
            const float* img = feat_kv + ((long)b * CC + c) * PP;
            kv[((long)b * CC + c) * (long)(KK * PP) + (long)k * PP + p] =
                f32_to_bf16(bilin_apply(img, cr));
        }
    }
}

// ---------------------------------------------------------------------------
// Per-pixel 9-point multi-head attention (K=9, HD=8): tiny -> VALU.
// ---------------------------------------------------------------------------
__global__ void attn_kernel(const ushort_t* __restrict__ qb,  // [B][96][P]
                            const ushort_t* __restrict__ kb,  // [B][96][9P]
                            const ushort_t* __restrict__ vb,  // [B][96][9P]
                            ushort_t* __restrict__ ob) {      // [B][96][P]
    long i = (long)blockIdx.x * blockDim.x + threadIdx.x;
    if (i >= (long)BB * NHH * PP) return;
    int p = (int)(i % PP); long t = i / PP;
    int h = (int)(t % NHH); int b = (int)(t / NHH);
    const float scale = 0.35355339059f;  // 1/sqrt(8)

    float q[HDD];
#pragma unroll
    for (int d = 0; d < HDD; ++d)
        q[d] = bf16_to_f32(qb[((long)b * CC + h * HDD + d) * PP + p]);

    float lg[KK]; float mx = -3.4e38f;
#pragma unroll
    for (int k = 0; k < KK; ++k) {
        float s = 0.f;
#pragma unroll
        for (int d = 0; d < HDD; ++d)
            s += q[d] * bf16_to_f32(
                kb[((long)b * CC + h * HDD + d) * (long)(KK * PP) + (long)k * PP + p]);
        lg[k] = s * scale;
        mx = fmaxf(mx, lg[k]);
    }
    float sum = 0.f;
#pragma unroll
    for (int k = 0; k < KK; ++k) { lg[k] = expf(lg[k] - mx); sum += lg[k]; }
    float inv = 1.f / sum;
#pragma unroll
    for (int d = 0; d < HDD; ++d) {
        float o = 0.f;
#pragma unroll
        for (int k = 0; k < KK; ++k)
            o += lg[k] * bf16_to_f32(
                vb[((long)b * CC + h * HDD + d) * (long)(KK * PP) + (long)k * PP + p]);
        ob[((long)b * CC + h * HDD + d) * PP + p] = f32_to_bf16(o * inv);
    }
}

// ---------------------------------------------------------------------------
// Host side
// ---------------------------------------------------------------------------
static long align256(long x) { return (x + 255) & ~255L; }

extern "C" void kernel_launch(void* const* d_in, const int* in_sizes, int n_in,
                              void* d_out, int out_size, void* d_ws, size_t ws_size,
                              hipStream_t stream) {
    const float* feat_t = (const float*)d_in[0];
    const float* feat_kv = (const float*)d_in[1];
    const float* flow = (const float*)d_in[2];
    const float* cw1 = (const float*)d_in[3];  const float* cb1 = (const float*)d_in[4];
    const float* a1  = (const float*)d_in[5];
    const float* cw2 = (const float*)d_in[6];  const float* cb2 = (const float*)d_in[7];
    const float* a2  = (const float*)d_in[8];
    const float* cw3 = (const float*)d_in[9];  const float* cb3 = (const float*)d_in[10];
    const float* wq = (const float*)d_in[11];  const float* bq = (const float*)d_in[12];
    const float* wk = (const float*)d_in[13];  const float* bk = (const float*)d_in[14];
    const float* wv = (const float*)d_in[15];  const float* bv = (const float*)d_in[16];
    const float* wo = (const float*)d_in[17];  const float* bo = (const float*)d_in[18];
    const float* wm1 = (const float*)d_in[19]; const float* bm1 = (const float*)d_in[20];
    const float* wm2 = (const float*)d_in[21]; const float* bm2 = (const float*)d_in[22];

    char* ws = (char*)d_ws;
    long cur = 0;
    auto alloc = [&](long bytes) -> char* {
        char* p = ws + cur;
        cur += align256(bytes);
        return p;
    };

    const long Pl = PP, NtotKV = (long)KK * PP;
    // packed weights
    ushort_t* pA1 = (ushort_t*)alloc(9L * 12 * 7 * 512 * 2);
    ushort_t* pA2 = (ushort_t*)alloc(9L * 6 * 6 * 512 * 2);
    ushort_t* pA3 = (ushort_t*)alloc(9L * 14 * 3 * 512 * 2);
    ushort_t* pQ  = (ushort_t*)alloc(6L * 3 * 512 * 2);
    ushort_t* pK  = (ushort_t*)alloc(6L * 3 * 512 * 2);
    ushort_t* pV  = (ushort_t*)alloc(6L * 3 * 512 * 2);
    ushort_t* pO  = (ushort_t*)alloc(6L * 3 * 512 * 2);
    ushort_t* pM1 = (ushort_t*)alloc(12L * 3 * 512 * 2);
    ushort_t* pM2 = (ushort_t*)alloc(6L * 6 * 512 * 2);
    // activations
    ushort_t* x1   = (ushort_t*)alloc((long)BB * 224 * Pl * 2);
    ushort_t* xtbf = (ushort_t*)alloc((long)BB * CC * Pl * 2);
    ushort_t* r1   = (ushort_t*)alloc((long)BB * 192 * Pl * 2);
    ushort_t* r2   = (ushort_t*)alloc((long)BB * 96 * Pl * 2);
    float*    offb = (float*)   alloc((long)BB * 216 * Pl * 4);
    ushort_t* kvb  = (ushort_t*)alloc((long)BB * CC * NtotKV * 2);
    ushort_t* kbuf = (ushort_t*)alloc((long)BB * CC * NtotKV * 2);
    ushort_t* vbuf = (ushort_t*)alloc((long)BB * CC * NtotKV * 2);
    ushort_t* qbuf = (ushort_t*)alloc((long)BB * CC * Pl * 2);
    ushort_t* obuf = (ushort_t*)alloc((long)BB * CC * Pl * 2);
    float*    xf32 = (float*)   alloc((long)BB * CC * Pl * 4);
    ushort_t* xbf  = (ushort_t*)alloc((long)BB * CC * Pl * 2);
    ushort_t* hbuf = (ushort_t*)alloc((long)BB * MLPC * Pl * 2);
    (void)ws_size; (void)in_sizes; (void)n_in; (void)out_size;

    auto cdiv = [](long a, long b) { return (int)((a + b - 1) / b); };
    auto pack = [&](const float* src, ushort_t* dst, int OC, int IC, int ICpad, int MT, int NTAP) {
        long tot = (long)NTAP * MT * (ICpad / 32) * 512;
        pack_w_kernel<<<cdiv(tot, 256), 256, 0, stream>>>(src, dst, OC, IC, ICpad, MT, NTAP);
    };
    pack(cw1, pA1, 192, 194, 224, 12, 9);
    pack(cw2, pA2,  96, 192, 192,  6, 9);
    pack(cw3, pA3, 216,  96,  96, 14, 9);
    pack(wq,  pQ,   96,  96,  96,  6, 1);
    pack(wk,  pK,   96,  96,  96,  6, 1);
    pack(wv,  pV,   96,  96,  96,  6, 1);
    pack(wo,  pO,   96,  96,  96,  6, 1);
    pack(wm1, pM1, 192,  96,  96, 12, 1);
    pack(wm2, pM2,  96, 192, 192,  6, 1);

    int ew = cdiv((long)BB * PP, 256);
    fill_x1_kernel<<<ew, 256, 0, stream>>>(feat_t, flow, x1, xtbf);
    warp_bilinear_kernel<<<ew, 256, 0, stream>>>(feat_kv, flow, x1);

    int cgrid = BB * HH * (WW / 64);  // 512
    conv3x3_wmma_kernel<<<cgrid, 256, 3 * 66 * (224 + 8) * 2, stream>>>(
        x1, pA1, cb1, a1, r1, nullptr, 224, 12, 192);
    conv3x3_wmma_kernel<<<cgrid, 256, 3 * 66 * (192 + 8) * 2, stream>>>(
        r1, pA2, cb2, a2, r2, nullptr, 192, 6, 96);
    conv3x3_wmma_kernel<<<cgrid, 256, 3 * 66 * (96 + 8) * 2, stream>>>(
        r2, pA3, cb3, nullptr, nullptr, offb, 96, 14, 216);

    deform_sample_kernel<<<cdiv((long)BB * GG * PP, 256), 256, 0, stream>>>(
        feat_kv, flow, offb, kvb);

    int g1 = (int)((long)BB * (Pl >> 6));       // 512
    int g9 = (int)((long)BB * (NtotKV >> 6));   // 4608
    gemm_wmma_kernel<<<g1, 256, 64 * (96 + 8) * 2, stream>>>(
        xtbf, pQ, bq, nullptr, qbuf, nullptr, 96, 6, 96, Pl, 0);
    gemm_wmma_kernel<<<g9, 256, 64 * (96 + 8) * 2, stream>>>(
        kvb, pK, bk, nullptr, kbuf, nullptr, 96, 6, 96, NtotKV, 0);
    gemm_wmma_kernel<<<g9, 256, 64 * (96 + 8) * 2, stream>>>(
        kvb, pV, bv, nullptr, vbuf, nullptr, 96, 6, 96, NtotKV, 0);

    attn_kernel<<<cdiv((long)BB * NHH * PP, 256), 256, 0, stream>>>(qbuf, kbuf, vbuf, obuf);

    gemm_wmma_kernel<<<g1, 256, 64 * (96 + 8) * 2, stream>>>(
        obuf, pO, bo, feat_t, xbf, xf32, 96, 6, 96, Pl, 0);
    gemm_wmma_kernel<<<g1, 256, 64 * (96 + 8) * 2, stream>>>(
        xbf, pM1, bm1, nullptr, hbuf, nullptr, 96, 12, 192, Pl, 1);
    gemm_wmma_kernel<<<g1, 256, 64 * (192 + 8) * 2, stream>>>(
        hbuf, pM2, bm2, xf32, nullptr, (float*)d_out, 192, 6, 96, Pl, 0);
}
